// MoELayer_12403865550894
// MI455X (gfx1250) — matrix operand order repaired and checked by reading, not compiled, not run on hardware
//
#include <hip/hip_runtime.h>
#include <stdint.h>

#define NTOK 4096
#define DIN  1024
#define HID  2048
#define OUTD 1024
#define NEXP 8
#define KSEL 2
#define NPAIR (NTOK * KSEL)

typedef __attribute__((ext_vector_type(16))) __bf16         v16bf;
typedef __attribute__((ext_vector_type(2)))  __bf16         v2bf;
typedef __attribute__((ext_vector_type(8)))  float          v8f;
typedef __attribute__((ext_vector_type(4)))  unsigned int   u32x4;

union Frag {
    v16bf v;
    u32x4 q[2];
};

__device__ __forceinline__ unsigned short f2bf(float f) {
    unsigned int u = __float_as_uint(f);
    u += 0x7FFFu + ((u >> 16) & 1u);           // round-to-nearest-even
    return (unsigned short)(u >> 16);
}

__device__ __forceinline__ unsigned int f2bf2(float lo, float hi) {
#if __has_builtin(__builtin_amdgcn_cvt_pk_bf16_f32)
    union { v2bf v; unsigned int u; } cv;
    cv.v = __builtin_amdgcn_cvt_pk_bf16_f32(lo, hi);
    return cv.u;
#else
    return (unsigned int)f2bf(lo) | ((unsigned int)f2bf(hi) << 16);
#endif
}

__device__ __forceinline__ int imin(int a, int b) { return a < b ? a : b; }

// ---------------------------------------------------------------------------
// Router: logits -> softmax -> top-2 -> normalized combine weights + compact
// per-expert token lists.
// ---------------------------------------------------------------------------
__global__ __launch_bounds__(256) void router_kernel(
    const float* __restrict__ x, const float* __restrict__ rw,
    const float* __restrict__ rb, int* __restrict__ counts,
    int* __restrict__ etok, int* __restrict__ tok_e, int* __restrict__ tok_s,
    float* __restrict__ tok_w)
{
    int n = blockIdx.x * blockDim.x + threadIdx.x;
    if (n >= NTOK) return;

    float acc[NEXP];
#pragma unroll
    for (int e = 0; e < NEXP; ++e) acc[e] = rb[e];

    const float* xr = x + (size_t)n * DIN;
    for (int d = 0; d < DIN; ++d) {
        float xv = xr[d];
        const float4* w4 = (const float4*)(rw + (size_t)d * NEXP);
        float4 a = w4[0], b = w4[1];
        acc[0] += xv * a.x; acc[1] += xv * a.y;
        acc[2] += xv * a.z; acc[3] += xv * a.w;
        acc[4] += xv * b.x; acc[5] += xv * b.y;
        acc[6] += xv * b.z; acc[7] += xv * b.w;
    }

    float mx = acc[0];
#pragma unroll
    for (int e = 1; e < NEXP; ++e) mx = fmaxf(mx, acc[e]);
    float p[NEXP], s = 0.f;
#pragma unroll
    for (int e = 0; e < NEXP; ++e) { p[e] = __expf(acc[e] - mx); s += p[e]; }
    float inv = 1.f / s;

    int e0 = 0; float p0 = p[0] * inv;
#pragma unroll
    for (int e = 1; e < NEXP; ++e) {
        float pe = p[e] * inv;
        if (pe > p0) { p0 = pe; e0 = e; }
    }
    int e1 = (e0 == 0) ? 1 : 0; float p1 = p[e1] * inv;
#pragma unroll
    for (int e = 0; e < NEXP; ++e) {
        if (e == e0) continue;
        float pe = p[e] * inv;
        if (pe > p1) { p1 = pe; e1 = e; }
    }
    float dnm = 1.f / (p0 + p1 + 1e-6f);
    float w0 = p0 * dnm, w1 = p1 * dnm;

    int s0 = atomicAdd(&counts[e0], 1);
    int s1 = atomicAdd(&counts[e1], 1);
    etok[e0 * NTOK + s0] = n;
    etok[e1 * NTOK + s1] = n;
    tok_e[n * 2 + 0] = e0; tok_e[n * 2 + 1] = e1;
    tok_s[n * 2 + 0] = s0; tok_s[n * 2 + 1] = s1;
    tok_w[n * 2 + 0] = w0; tok_w[n * 2 + 1] = w1;
}

__global__ void scan_kernel(const int* __restrict__ counts,
                            int* __restrict__ segoff)
{
    if (threadIdx.x == 0 && blockIdx.x == 0) {
        int off = 0;
        for (int e = 0; e < NEXP; ++e) { segoff[e] = off; off += counts[e]; }
        segoff[NEXP] = off;
    }
}

// ---------------------------------------------------------------------------
// Tiled bf16 WMMA GEMM, one kernel for all 3 expert-FFN stages.
// Block: 256 threads (8 waves). Tile: 64(M) x 64(N), k-step 64.
// KD/NW are compile-time so global-load strides fold into 24-bit ioffsets.
//   STAGE 1: A = gathered x rows (f32->bf16), out = relu(.+b1) -> bf16 h1
//   STAGE 2: A = h1 (bf16),                   out = relu(.+b2) -> bf16 h2
//   STAGE 3: A = h2 (bf16),                   out = (.+b3)     -> f32 y
// ---------------------------------------------------------------------------
template <int STAGE, int KD, int NW>
__global__ __launch_bounds__(256) void moe_gemm(
    const float* __restrict__ W,      // [E][KD][NW]
    const float* __restrict__ Bias,   // [E][NW]
    const float* __restrict__ X,      // stage1 only: [NTOK][DIN]
    const unsigned short* __restrict__ Ain, // stage2/3: [NPAIR][KD] bf16
    unsigned short* __restrict__ OutB,      // stage1/2: [NPAIR][NW] bf16
    float* __restrict__ OutF,               // stage3:   [NPAIR][NW] f32
    const int* __restrict__ counts, const int* __restrict__ segoff,
    const int* __restrict__ etok)
{
    const int e   = blockIdx.z;
    const int cnt = counts[e];
    const int m0  = blockIdx.x * 64;
    if (m0 >= cnt) return;
    const int seg = segoff[e];
    const int n0  = blockIdx.y * 64;

    const float* Wb = W + (size_t)e * KD * NW;

    __shared__ unsigned short lA[64][72];   // [m][k] bf16, row stride 144B
    __shared__ unsigned short lB[64][72];   // [n][k] bf16 (B transposed)

    const int tid  = threadIdx.x;
    const int lane = tid & 31;
    const int wid  = tid >> 5;
    const int wN   = wid & 3;
    const int wMb  = (wid >> 2) * 32;
    const int half = lane >> 4;
    const int l15  = lane & 15;

    // A staging: row = tid>>2 (0..63), k-chunks (tid&3)*8 and +32
    const int ar  = tid >> 2;
    const int ak0 = (tid & 3) * 8;
    const float* ag = nullptr;          // stage1: gathered x row base
    const unsigned short* ah = nullptr; // stage2/3: bf16 row base
    if (STAGE == 1) {
        int atok = etok[e * NTOK + imin(m0 + ar, cnt - 1)];
        ag = X + (size_t)atok * DIN + ak0;
    } else {
        ah = Ain + (size_t)imin(seg + m0 + ar, NPAIR - 1) * KD + ak0;
    }
    // B staging: n = tid&63, k-chunks (tid>>6)*8 and +32 (loads run along k,
    // coalesced across the 64 lanes sharing a k-group; one b128 LDS store).
    const int bn  = tid & 63;
    const int bk0 = (tid >> 6) * 8;
    const float* bg = Wb + (size_t)bk0 * NW + n0 + bn;

    v8f accs[2] = {{}, {}};

    for (int kt = 0; kt < KD; kt += 64) {
        // ---- stage A tile (64x64) into LDS as bf16 ----
#pragma unroll
        for (int kh = 0; kh < 2; ++kh) {
            const int k0 = ak0 + kh * 32;
            if (STAGE == 1) {
                const float4 v0 = *(const float4*)(ag + kt + kh * 32);
                const float4 v1 = *(const float4*)(ag + kt + kh * 32 + 4);
                u32x4 hh;
                hh.x = f2bf2(v0.x, v0.y); hh.y = f2bf2(v0.z, v0.w);
                hh.z = f2bf2(v1.x, v1.y); hh.w = f2bf2(v1.z, v1.w);
                *(u32x4*)&lA[ar][k0] = hh;
            } else {
                *(u32x4*)&lA[ar][k0] = *(const u32x4*)(ah + kt + kh * 32);
            }
        }
        // ---- stage B tile (64k x 64n) into LDS transposed to [n][k] ----
#pragma unroll
        for (int kh = 0; kh < 2; ++kh) {
            const float* bp = bg + ((size_t)kt + kh * 32) * NW;
            float f[8];
#pragma unroll
            for (int j = 0; j < 8; ++j)
                f[j] = bp[(size_t)j * NW];   // imm offsets: j*NW*4 < 2^23
            u32x4 hh;
            hh.x = f2bf2(f[0], f[1]); hh.y = f2bf2(f[2], f[3]);
            hh.z = f2bf2(f[4], f[5]); hh.w = f2bf2(f[6], f[7]);
            *(u32x4*)&lB[bn][bk0 + kh * 32] = hh;
        }
        if (kt + 64 < KD)
            __builtin_prefetch(bg + ((size_t)kt + 64) * NW, 0, 1);
        __syncthreads();

        // ---- 4 wmmas per wave per k-step ----
#pragma unroll
        for (int ks = 0; ks < 64; ks += 32) {
            Frag a0, a1, b;
            const int ab = ks + 8 * half;   // A: lane<16 K 0..7/16..23
            a0.q[0] = *(const u32x4*)&lA[wMb + l15][ab];
            a0.q[1] = *(const u32x4*)&lA[wMb + l15][ab + 16];
            a1.q[0] = *(const u32x4*)&lA[wMb + 16 + l15][ab];
            a1.q[1] = *(const u32x4*)&lA[wMb + 16 + l15][ab + 16];
            const int bb = ks + 8 * half;   // B: lane<16 K 0..7/16..23
            b.q[0] = *(const u32x4*)&lB[wN * 16 + l15][bb];
            b.q[1] = *(const u32x4*)&lB[wN * 16 + l15][bb + 16];
            accs[0] = __builtin_amdgcn_wmma_f32_16x16x32_bf16(
                false, a0.v, false, b.v, (short)0, accs[0], false, false);
            accs[1] = __builtin_amdgcn_wmma_f32_16x16x32_bf16(
                false, a1.v, false, b.v, (short)0, accs[1], false, false);
        }
        __syncthreads();
    }

    // ---- epilogue ----
    const int n_g = n0 + wN * 16 + l15;
    const float bv = Bias[(size_t)e * NW + n_g];
#pragma unroll
    for (int t = 0; t < 2; ++t) {
        const int rb = m0 + wMb + t * 16 + 8 * half;  // C: VGPR r -> M=r(+8)
#pragma unroll
        for (int r = 0; r < 8; ++r) {
            const int row = rb + r;
            if (row < cnt) {
                float v = accs[t][r] + bv;
                if (STAGE < 3) {
                    v = fmaxf(v, 0.f);
                    OutB[(size_t)(seg + row) * NW + n_g] = f2bf(v);
                } else {
                    OutF[(size_t)(seg + row) * NW + n_g] = v;
                }
            }
        }
    }
}

// ---------------------------------------------------------------------------
// Combine: out[n,:] = sum_k w_k * y[segoff[e_k]+slot_k, :]   (no atomics)
// ---------------------------------------------------------------------------
__global__ __launch_bounds__(256) void combine_kernel(
    const float* __restrict__ ybuf, const int* __restrict__ tok_e,
    const int* __restrict__ tok_s, const float* __restrict__ tok_w,
    const int* __restrict__ segoff, float* __restrict__ out)
{
    int idx = blockIdx.x * blockDim.x + threadIdx.x;  // NTOK * OUTD/4
    int n  = idx / (OUTD / 4);
    int o4 = (idx % (OUTD / 4)) * 4;
    if (n >= NTOK) return;

    float4 r = make_float4(0.f, 0.f, 0.f, 0.f);
#pragma unroll
    for (int k = 0; k < KSEL; ++k) {
        int e = tok_e[n * 2 + k];
        int p = segoff[e] + tok_s[n * 2 + k];
        float w = tok_w[n * 2 + k];
        const float4 y = *(const float4*)&ybuf[(size_t)p * OUTD + o4];
        r.x += w * y.x; r.y += w * y.y; r.z += w * y.z; r.w += w * y.w;
    }
    *(float4*)&out[(size_t)n * OUTD + o4] = r;
}

// ---------------------------------------------------------------------------
extern "C" void kernel_launch(void* const* d_in, const int* in_sizes, int n_in,
                              void* d_out, int out_size, void* d_ws,
                              size_t ws_size, hipStream_t stream)
{
    (void)in_sizes; (void)n_in; (void)out_size; (void)ws_size;

    const float* x  = (const float*)d_in[0];
    const float* rw = (const float*)d_in[1];
    const float* rb = (const float*)d_in[2];
    const float* w1 = (const float*)d_in[3];
    const float* b1 = (const float*)d_in[4];
    const float* w2 = (const float*)d_in[5];
    const float* b2 = (const float*)d_in[6];
    const float* w3 = (const float*)d_in[7];
    const float* b3 = (const float*)d_in[8];
    float* out = (float*)d_out;

    char* p = (char*)d_ws;
    int* counts = (int*)p;  p += 256;
    int* segoff = (int*)p;  p += 256;
    int* etok   = (int*)p;  p += (size_t)NEXP * NTOK * sizeof(int);
    int* tok_e  = (int*)p;  p += (size_t)NPAIR * sizeof(int);
    int* tok_s  = (int*)p;  p += (size_t)NPAIR * sizeof(int);
    float* tok_w = (float*)p; p += (size_t)NPAIR * sizeof(float);
    p = (char*)(((uintptr_t)p + 255) & ~(uintptr_t)255);
    unsigned short* h1 = (unsigned short*)p; p += (size_t)NPAIR * HID * 2;
    unsigned short* h2 = (unsigned short*)p; p += (size_t)NPAIR * HID * 2;
    float* ybuf = (float*)p;                 p += (size_t)NPAIR * OUTD * 4;

    hipMemsetAsync(counts, 0, NEXP * sizeof(int), stream);

    router_kernel<<<NTOK / 256, 256, 0, stream>>>(x, rw, rb, counts, etok,
                                                  tok_e, tok_s, tok_w);
    scan_kernel<<<1, 1, 0, stream>>>(counts, segoff);

    dim3 blk(256);
    moe_gemm<1, DIN, HID><<<dim3(NTOK / 64, HID / 64, NEXP), blk, 0, stream>>>(
        w1, b1, x, nullptr, h1, nullptr, counts, segoff, etok);
    moe_gemm<2, HID, HID><<<dim3(NTOK / 64, HID / 64, NEXP), blk, 0, stream>>>(
        w2, b2, nullptr, h1, h2, nullptr, counts, segoff, etok);
    moe_gemm<3, HID, OUTD><<<dim3(NTOK / 64, OUTD / 64, NEXP), blk, 0, stream>>>(
        w3, b3, nullptr, h2, nullptr, ybuf, counts, segoff, etok);

    combine_kernel<<<(NTOK * (OUTD / 4)) / 256, 256, 0, stream>>>(
        ybuf, tok_e, tok_s, tok_w, segoff, out);
}
